// MultiHeadAttentionBlock_14671608283228
// MI455X (gfx1250) — compile-verified, matrix operand order checked by reading, not run on hardware
//
#include <hip/hip_runtime.h>
#include <hip/hip_bf16.h>

// ---------------------------------------------------------------------------
// MHA block for MI455X (gfx1250): bf16 WMMA, flash-attention core,
// async global->LDS staging for the V operand.
// ---------------------------------------------------------------------------

#define BATCH   2
#define SEQ     2048
#define DMODEL  1024
#define NHEADS  16
#define DK      64
#define TOKENS  (BATCH * SEQ)          // 4096
#define SCALE_F 0.125f                 // 1/sqrt(64)
#define NEG_BIG (-1.0e9f)

typedef __bf16 bf16;
typedef __attribute__((ext_vector_type(16))) __bf16 v16bf;
typedef __attribute__((ext_vector_type(8)))  __bf16 v8bf;
typedef __attribute__((ext_vector_type(8)))  float  v8f;
typedef __attribute__((ext_vector_type(4)))  float  v4f;

// A/B fragment half-row: elements 0..7 at base, 8..15 at base+16
// (K = (e<8?e:e+8)+khalf pattern; caller pre-adds khalf). Two b128 loads.
__device__ __forceinline__ v16bf make_frag(const bf16* __restrict__ base) {
    v8bf lo = *reinterpret_cast<const v8bf*>(base);
    v8bf hi = *reinterpret_cast<const v8bf*>(base + 16);
    v16bf f;
#pragma unroll
    for (int i = 0; i < 8; ++i) { f[i] = lo[i]; f[i + 8] = hi[i]; }
    return f;
}

__device__ __forceinline__ v8f wmma_bf16(v16bf a, v16bf b, v8f c) {
    return __builtin_amdgcn_wmma_f32_16x16x32_bf16(false, a, false, b,
                                                   (short)0, c, false, false);
}

// gfx1250 async data mover: 16B global -> LDS, tracked by ASYNCcnt.
__device__ __forceinline__ void async_copy_b128(unsigned lds_off,
                                                const void* gaddr) {
    asm volatile("global_load_async_to_lds_b128 %0, %1, off"
                 :: "v"(lds_off), "v"(gaddr) : "memory");
}
__device__ __forceinline__ void wait_asynccnt0() {
    asm volatile("s_wait_asynccnt 0x0" ::: "memory");
}
__device__ __forceinline__ void wait_dscnt0() {
    asm volatile("s_wait_dscnt 0x0" ::: "memory");
}

// ---------------------------------------------------------------------------
// fp32 -> bf16 elementwise cast
// ---------------------------------------------------------------------------
__global__ void cast_bf16_kernel(const float* __restrict__ x,
                                 bf16* __restrict__ y, int n) {
    int i = blockIdx.x * blockDim.x + threadIdx.x;
    if (i < n) y[i] = (bf16)x[i];
}

// W[K][N] (fp32) -> Wt[N][K] (bf16); consecutive threads write consecutive k.
__global__ void transpose_cast_kernel(const float* __restrict__ W,
                                      bf16* __restrict__ Wt, int K, int N) {
    size_t i = (size_t)blockIdx.x * blockDim.x + threadIdx.x;
    if (i < (size_t)K * N) {
        int n = (int)(i / K);
        int k = (int)(i % K);
        Wt[i] = (bf16)W[(size_t)k * N + n];
    }
}

// ---------------------------------------------------------------------------
// C[M,N] = A[M,K] @ Bt[N,K]^T + bias.
// One wave computes a 16x64 strip: 1 A frag feeds 4 WMMAs per K-step,
// software-pipelined so loads for K+32 overlap WMMAs for K.
// ---------------------------------------------------------------------------
__global__ __launch_bounds__(128)
void gemm_bf16_kernel(const bf16* __restrict__ A, const bf16* __restrict__ Bt,
                      const float* __restrict__ bias,
                      bf16* __restrict__ out_bf, float* __restrict__ out_f32,
                      int M, int N, int K) {
    const int lane   = threadIdx.x & 31;
    const int wave   = threadIdx.x >> 5;
    const int mtile  = blockIdx.x;
    const int nstrip = blockIdx.y * 4 + wave;      // 64-column strip
    const int rlane  = lane & 15;
    const int khalf  = (lane < 16) ? 0 : 8;

    const size_t bstride = (size_t)16 * K;         // one 16-col tile of Bt
    const bf16* arow = A  + (size_t)(mtile * 16 + rlane) * K + khalf;
    const bf16* brow = Bt + (size_t)(nstrip * 64 + rlane) * K + khalf;

    v8f acc0 = {}, acc1 = {}, acc2 = {}, acc3 = {};

    v16bf aC  = make_frag(arow);
    v16bf bC0 = make_frag(brow);
    v16bf bC1 = make_frag(brow + bstride);
    v16bf bC2 = make_frag(brow + 2 * bstride);
    v16bf bC3 = make_frag(brow + 3 * bstride);

    int kb = 0;
    for (; kb + 32 < K; kb += 32) {
        __builtin_prefetch(arow + kb + 96, 0, 0);
        v16bf aN  = make_frag(arow + kb + 32);
        v16bf bN0 = make_frag(brow + kb + 32);
        v16bf bN1 = make_frag(brow + bstride + kb + 32);
        v16bf bN2 = make_frag(brow + 2 * bstride + kb + 32);
        v16bf bN3 = make_frag(brow + 3 * bstride + kb + 32);

        acc0 = wmma_bf16(aC, bC0, acc0);
        acc1 = wmma_bf16(aC, bC1, acc1);
        acc2 = wmma_bf16(aC, bC2, acc2);
        acc3 = wmma_bf16(aC, bC3, acc3);

        aC = aN; bC0 = bN0; bC1 = bN1; bC2 = bN2; bC3 = bN3;
    }
    acc0 = wmma_bf16(aC, bC0, acc0);
    acc1 = wmma_bf16(aC, bC1, acc1);
    acc2 = wmma_bf16(aC, bC2, acc2);
    acc3 = wmma_bf16(aC, bC3, acc3);

    const int mbase = mtile * 16 + ((lane < 16) ? 0 : 8);
    v8f accs[4] = {acc0, acc1, acc2, acc3};
#pragma unroll
    for (int t = 0; t < 4; ++t) {
        const int ncol = nstrip * 64 + t * 16 + rlane;
        const float bv = bias ? bias[ncol] : 0.0f;
#pragma unroll
        for (int r = 0; r < 8; ++r) {
            float vv = accs[t][r] + bv;
            size_t idx = (size_t)(mbase + r) * N + ncol;
            if (out_f32) out_f32[idx] = vv;
            if (out_bf)  out_bf[idx]  = (bf16)vv;
        }
    }
}

// ---------------------------------------------------------------------------
// Flash attention: one wave per (b, h, 16-query tile); stream 32-key blocks.
// V block is staged to LDS with global_load_async_to_lds_b128 and the wait is
// deferred until after QK^T + softmax, overlapping DMA with WMMA/VALU work.
// ---------------------------------------------------------------------------
__global__ __launch_bounds__(32)
void attn_kernel(const bf16* __restrict__ Q, const bf16* __restrict__ Km,
                 const bf16* __restrict__ V, const int* __restrict__ mask,
                 bf16* __restrict__ ctx) {
    __shared__ __align__(16) float pLDS[16 * 32];   // P tile, f32
    __shared__ __align__(16) bf16  vLDS[32 * 64];   // staged V block

    const int lane  = threadIdx.x & 31;
    const int qt    = blockIdx.x % (SEQ / 16);
    const int h     = (blockIdx.x / (SEQ / 16)) % NHEADS;
    const int b     = blockIdx.x / ((SEQ / 16) * NHEADS);
    const int rlane = lane & 15;
    const int khalf = (lane < 16) ? 0 : 8;
    const int hrow  = (lane < 16) ? 0 : 8;

    // Q tile A-fragments (dk 0..31 and 32..63)
    const size_t qoff = ((size_t)(b * SEQ) + qt * 16 + rlane) * DMODEL + h * DK;
    const v16bf qf0 = make_frag(Q + qoff + khalf);
    const v16bf qf1 = make_frag(Q + qoff + 32 + khalf);

    v8f o0 = {}, o1 = {}, o2 = {}, o3 = {};
    float mrun[8], lrun[8];
#pragma unroll
    for (int r = 0; r < 8; ++r) { mrun[r] = -INFINITY; lrun[r] = 0.0f; }

    const size_t maskbase = (size_t)b * SEQ * SEQ;
    const unsigned vbase  = (unsigned)(size_t)&vLDS[0];

    for (int kb = 0; kb < SEQ; kb += 32) {
        // ---- WAR guard: prior iteration's DS reads of vLDS/pLDS must drain
        //      before the async engine may overwrite vLDS ----
        wait_dscnt0();

        // ---- async-stage V block (32 keys x 64 dk): 8 x b128 per lane ----
        {
            const bf16* gsrc =
                V + ((size_t)(b * SEQ) + kb + lane) * DMODEL + h * DK;
            const unsigned lrow = vbase + (unsigned)(lane * 64) * sizeof(bf16);
#pragma unroll
            for (int j = 0; j < 8; ++j)
                async_copy_b128(lrow + j * 16, gsrc + j * 8);
        }

        // ---- scores: two 16x16 tiles, contraction over dk=64 ----
        v8f s0 = {}, s1 = {};
        {
            const size_t k0 = ((size_t)(b * SEQ) + kb + rlane) * DMODEL + h * DK;
            const size_t k1 = ((size_t)(b * SEQ) + kb + 16 + rlane) * DMODEL + h * DK;
            s0 = wmma_bf16(qf0, make_frag(Km + k0 + khalf),      s0);
            s0 = wmma_bf16(qf1, make_frag(Km + k0 + 32 + khalf), s0);
            s1 = wmma_bf16(qf0, make_frag(Km + k1 + khalf),      s1);
            s1 = wmma_bf16(qf1, make_frag(Km + k1 + 32 + khalf), s1);
        }

        // ---- scale + mask ----
        float x0[8], x1[8], tred[8];
#pragma unroll
        for (int r = 0; r < 8; ++r) {
            const int qrow = qt * 16 + r + hrow;
            const int* mrow = mask + maskbase + (size_t)qrow * SEQ + kb + rlane;
            float a = s0[r] * SCALE_F; if (mrow[0]  == 0) a = NEG_BIG;
            float c = s1[r] * SCALE_F; if (mrow[16] == 0) c = NEG_BIG;
            x0[r] = a; x1[r] = c;
            tred[r] = fmaxf(a, c);
        }
        // row max across the 16-lane half (xor 8,4,2,1 stays inside half)
#pragma unroll
        for (int off = 8; off >= 1; off >>= 1)
#pragma unroll
            for (int r = 0; r < 8; ++r)
                tred[r] = fmaxf(tred[r], __shfl_xor(tred[r], off, 32));

        // ---- online softmax update ----
        float alpha[8];
#pragma unroll
        for (int r = 0; r < 8; ++r) {
            const float mn = fmaxf(mrun[r], tred[r]);
            alpha[r] = __expf(mrun[r] - mn);
            mrun[r]  = mn;
            x0[r] = __expf(x0[r] - mn);
            x1[r] = __expf(x1[r] - mn);
        }
        float rs[8];
#pragma unroll
        for (int r = 0; r < 8; ++r) rs[r] = x0[r] + x1[r];
#pragma unroll
        for (int off = 8; off >= 1; off >>= 1)
#pragma unroll
            for (int r = 0; r < 8; ++r)
                rs[r] += __shfl_xor(rs[r], off, 32);
#pragma unroll
        for (int r = 0; r < 8; ++r) lrun[r] = lrun[r] * alpha[r] + rs[r];
#pragma unroll
        for (int r = 0; r < 8; ++r) {
            o0[r] *= alpha[r]; o1[r] *= alpha[r];
            o2[r] *= alpha[r]; o3[r] *= alpha[r];
        }

        // ---- C-layout P -> LDS (DS pipe is in-order per wave) ----
#pragma unroll
        for (int r = 0; r < 8; ++r) {
            pLDS[(r + hrow) * 32 + rlane]      = x0[r];
            pLDS[(r + hrow) * 32 + rlane + 16] = x1[r];
        }

        // ---- reload P as A-layout bf16 fragment, vectorized (4 x b128) ----
        v16bf pf;
        {
            const float* prow = &pLDS[rlane * 32 + khalf];
            v4f p0 = *reinterpret_cast<const v4f*>(prow);
            v4f p1 = *reinterpret_cast<const v4f*>(prow + 4);
            v4f p2 = *reinterpret_cast<const v4f*>(prow + 16);
            v4f p3 = *reinterpret_cast<const v4f*>(prow + 20);
#pragma unroll
            for (int i = 0; i < 4; ++i) {
                pf[i]      = (bf16)p0[i];
                pf[i + 4]  = (bf16)p1[i];
                pf[i + 8]  = (bf16)p2[i];
                pf[i + 12] = (bf16)p3[i];
            }
        }

        // ---- V DMA must be complete before reading vLDS ----
        wait_asynccnt0();

        // ---- V B-fragments from LDS (4 dk tiles) ----
        v16bf vf0, vf1, vf2, vf3;
#pragma unroll
        for (int e = 0; e < 16; ++e) {
            const int key = ((e < 8) ? e : e + 8) + khalf;
            vf0[e] = vLDS[key * 64 +  0 + rlane];
            vf1[e] = vLDS[key * 64 + 16 + rlane];
            vf2[e] = vLDS[key * 64 + 32 + rlane];
            vf3[e] = vLDS[key * 64 + 48 + rlane];
        }

        o0 = wmma_bf16(pf, vf0, o0);
        o1 = wmma_bf16(pf, vf1, o1);
        o2 = wmma_bf16(pf, vf2, o2);
        o3 = wmma_bf16(pf, vf3, o3);
    }

    // ---- finalize: divide by row sum, store ctx in [B,S,H*Dk] bf16 ----
#pragma unroll
    for (int r = 0; r < 8; ++r) {
        const float inv  = (lrun[r] > 0.0f) ? (1.0f / lrun[r]) : 0.0f;
        const int   mrow = qt * 16 + r + hrow;
        const size_t ob  = ((size_t)(b * SEQ) + mrow) * DMODEL + h * DK + rlane;
        ctx[ob +  0] = (bf16)(o0[r] * inv);
        ctx[ob + 16] = (bf16)(o1[r] * inv);
        ctx[ob + 32] = (bf16)(o2[r] * inv);
        ctx[ob + 48] = (bf16)(o3[r] * inv);
    }
}

// ---------------------------------------------------------------------------
// Host side
// ---------------------------------------------------------------------------
extern "C" void kernel_launch(void* const* d_in, const int* in_sizes, int n_in,
                              void* d_out, int out_size, void* d_ws, size_t ws_size,
                              hipStream_t stream) {
    (void)in_sizes; (void)n_in; (void)out_size; (void)ws_size;

    const float* q    = (const float*)d_in[0];
    const float* k    = (const float*)d_in[1];
    const float* v    = (const float*)d_in[2];
    const int*   mask = (const int*)  d_in[3];
    const float* Wq   = (const float*)d_in[4];
    const float* bq   = (const float*)d_in[5];
    const float* Wk   = (const float*)d_in[6];
    const float* bk   = (const float*)d_in[7];
    const float* Wv   = (const float*)d_in[8];
    const float* bv   = (const float*)d_in[9];
    const float* Wo   = (const float*)d_in[10];
    const float* bo   = (const float*)d_in[11];

    const size_t xelems = (size_t)TOKENS * DMODEL;
    const size_t welems = (size_t)DMODEL * DMODEL;

    char* ws = (char*)d_ws;
    size_t off = 0;
    auto carve = [&](size_t bytes) -> char* {
        char* p = ws + off;
        off += (bytes + 255) & ~(size_t)255;
        return p;
    };

    bf16* qb   = (bf16*)carve(xelems * sizeof(bf16));
    bf16* kb_  = (bf16*)carve(xelems * sizeof(bf16));
    bf16* vb   = (bf16*)carve(xelems * sizeof(bf16));
    bf16* wqt  = (bf16*)carve(welems * sizeof(bf16));
    bf16* wkt  = (bf16*)carve(welems * sizeof(bf16));
    bf16* wvt  = (bf16*)carve(welems * sizeof(bf16));
    bf16* wot  = (bf16*)carve(welems * sizeof(bf16));
    bf16* Qp   = (bf16*)carve(xelems * sizeof(bf16));
    bf16* Kp   = (bf16*)carve(xelems * sizeof(bf16));
    bf16* Vp   = (bf16*)carve(xelems * sizeof(bf16));
    bf16* ctxb = (bf16*)carve(xelems * sizeof(bf16));

    {
        const int n = (int)xelems;
        const int g = (n + 255) / 256;
        cast_bf16_kernel<<<g, 256, 0, stream>>>(q, qb, n);
        cast_bf16_kernel<<<g, 256, 0, stream>>>(k, kb_, n);
        cast_bf16_kernel<<<g, 256, 0, stream>>>(v, vb, n);
    }
    {
        const int g = (int)((welems + 255) / 256);
        transpose_cast_kernel<<<g, 256, 0, stream>>>(Wq, wqt, DMODEL, DMODEL);
        transpose_cast_kernel<<<g, 256, 0, stream>>>(Wk, wkt, DMODEL, DMODEL);
        transpose_cast_kernel<<<g, 256, 0, stream>>>(Wv, wvt, DMODEL, DMODEL);
        transpose_cast_kernel<<<g, 256, 0, stream>>>(Wo, wot, DMODEL, DMODEL);
    }

    // Projections: one wave = 16x64 strip; grid (M/16, N/(4*64))
    dim3 gg(TOKENS / 16, DMODEL / 256);   // (256, 4), 4 waves/block
    gemm_bf16_kernel<<<gg, 128, 0, stream>>>(qb,  wqt, bq, Qp, nullptr,
                                             TOKENS, DMODEL, DMODEL);
    gemm_bf16_kernel<<<gg, 128, 0, stream>>>(kb_, wkt, bk, Kp, nullptr,
                                             TOKENS, DMODEL, DMODEL);
    gemm_bf16_kernel<<<gg, 128, 0, stream>>>(vb,  wvt, bv, Vp, nullptr,
                                             TOKENS, DMODEL, DMODEL);

    attn_kernel<<<BATCH * NHEADS * (SEQ / 16), 32, 0, stream>>>(
        Qp, Kp, Vp, mask, ctxb);

    gemm_bf16_kernel<<<gg, 128, 0, stream>>>(ctxb, wot, bo, nullptr,
                                             (float*)d_out,
                                             TOKENS, DMODEL, DMODEL);
}
